// ReflectiveCore_44074954392035
// MI455X (gfx1250) — matrix operand order repaired and checked by reading, not compiled, not run on hardware
//
#include <hip/hip_runtime.h>

typedef __attribute__((ext_vector_type(16))) __bf16        v16bf;
typedef __attribute__((ext_vector_type(8)))  float         v8f;
typedef __attribute__((ext_vector_type(4)))  unsigned int  u32x4;

union AF { v16bf v; u32x4 q[2]; };

constexpr int V_   = 1000;
constexpr int S_   = 8;
constexpr int D_   = 512;
constexpr int H_   = 8;
constexpr int NL_  = 2;
constexpr int DFF_ = 2048;
constexpr int B_   = 8;
constexpr int T_   = 1024;
constexpr int C_   = 16;
constexpr int L_   = C_ + T_;      // 1040
constexpr int M_   = B_ * L_;      // 8320
constexpr int NQT_ = L_ / 16;      // 65

__device__ __forceinline__ unsigned short f2bf(float f) {
  unsigned u = __float_as_uint(f);
  u += 0x7FFFu + ((u >> 16) & 1u);      // round-to-nearest-even
  return (unsigned short)(u >> 16);
}

__device__ __forceinline__ v8f wmma_bf16(const AF& a, const AF& b, v8f c) {
  return __builtin_amdgcn_wmma_f32_16x16x32_bf16(
      /*neg_a=*/false, a.v, /*neg_b=*/false, b.v,
      /*c_mod=*/(short)0, c, /*reuse_a=*/false, /*reuse_b=*/false);
}

// ---------------------------------------------------------------- converts
__global__ void f32_to_bf16_kernel(const float* __restrict__ src,
                                   unsigned short* __restrict__ dst,
                                   long long n) {
  long long i = (long long)blockIdx.x * blockDim.x + threadIdx.x;
  long long stride = (long long)gridDim.x * blockDim.x;
  for (; i < n; i += stride) dst[i] = f2bf(src[i]);
}

// ---------------------------------------------------------------- embedding
// one block per row m = b*L + l ; builds x = [prefix ; gather(emb)+b_in] + pos
__global__ void embed_kernel(const int* __restrict__ token,
                             const int* __restrict__ state,
                             const float* __restrict__ prefix,
                             const float* __restrict__ emb,
                             const float* __restrict__ b_in,
                             const float* __restrict__ pos,
                             float* __restrict__ xf,
                             unsigned short* __restrict__ xb,
                             unsigned short* __restrict__ memb) {
  int m = blockIdx.x;
  int b = m / L_;
  int l = m - b * L_;
  for (int d = threadIdx.x; d < D_; d += blockDim.x) {
    float v;
    if (l < C_) {
      v = prefix[((size_t)b * C_ + l) * D_ + d];
    } else {
      int t = l - C_;
      int idx = token[b * T_ + t] * S_ + state[b];
      v = emb[(size_t)idx * D_ + d] + b_in[d];
    }
    v += pos[(size_t)l * D_ + d];
    size_t o = (size_t)m * D_ + d;
    xf[o] = v;
    unsigned short h = f2bf(v);
    xb[o] = h;
    memb[o] = h;
  }
}

// ---------------------------------------------------------------- WMMA GEMM
// A  : bf16 [M,K] row-major (activations)
// W  : bf16 [N,K] row-major (weight, out = A @ W^T)
// Cf : optional f32 [M,N] ; Cb : optional bf16 [M,N] ; bias optional f32[N]
// each wave -> 32x64 tile (2x4 of 16x16), 4 waves per block along M:
// 8 WMMAs per 12 b128 fragment loads, long independent WMMA chains.
__global__ __launch_bounds__(128)
void gemm_kernel(const unsigned short* __restrict__ A,
                 const unsigned short* __restrict__ W,
                 const float* __restrict__ bias,
                 float* __restrict__ Cf,
                 unsigned short* __restrict__ Cb,
                 int Mdim, int Ndim, int Kdim, int relu) {
  int lane = threadIdx.x & 31;
  int wid  = threadIdx.x >> 5;
  int n = lane & 15;
  int g = lane >> 4;
  int tm = (blockIdx.y * 4 + wid) * 32;
  int tn = blockIdx.x * 64;
  if (tm >= Mdim) return;

  const unsigned short* arow0 = A + (size_t)(tm + n) * Kdim;
  const unsigned short* arow1 = A + (size_t)(tm + 16 + n) * Kdim;
  const unsigned short* brow0 = W + (size_t)(tn + n) * Kdim;
  const unsigned short* brow1 = W + (size_t)(tn + 16 + n) * Kdim;
  const unsigned short* brow2 = W + (size_t)(tn + 32 + n) * Kdim;
  const unsigned short* brow3 = W + (size_t)(tn + 48 + n) * Kdim;

  v8f acc[2][4] = {};
  for (int kk = 0; kk < Kdim; kk += 32) {
    AF a0, a1, b0, b1, b2, b3;
    a0.q[0] = *(const u32x4*)(arow0 + kk + g * 16);
    a0.q[1] = *(const u32x4*)(arow0 + kk + g * 16 + 8);
    a1.q[0] = *(const u32x4*)(arow1 + kk + g * 16);
    a1.q[1] = *(const u32x4*)(arow1 + kk + g * 16 + 8);
    b0.q[0] = *(const u32x4*)(brow0 + kk + g * 16);
    b0.q[1] = *(const u32x4*)(brow0 + kk + g * 16 + 8);
    b1.q[0] = *(const u32x4*)(brow1 + kk + g * 16);
    b1.q[1] = *(const u32x4*)(brow1 + kk + g * 16 + 8);
    b2.q[0] = *(const u32x4*)(brow2 + kk + g * 16);
    b2.q[1] = *(const u32x4*)(brow2 + kk + g * 16 + 8);
    b3.q[0] = *(const u32x4*)(brow3 + kk + g * 16);
    b3.q[1] = *(const u32x4*)(brow3 + kk + g * 16 + 8);
    acc[0][0] = wmma_bf16(a0, b0, acc[0][0]);
    acc[0][1] = wmma_bf16(a0, b1, acc[0][1]);
    acc[0][2] = wmma_bf16(a0, b2, acc[0][2]);
    acc[0][3] = wmma_bf16(a0, b3, acc[0][3]);
    acc[1][0] = wmma_bf16(a1, b0, acc[1][0]);
    acc[1][1] = wmma_bf16(a1, b1, acc[1][1]);
    acc[1][2] = wmma_bf16(a1, b2, acc[1][2]);
    acc[1][3] = wmma_bf16(a1, b3, acc[1][3]);
  }

  float bv[4];
#pragma unroll
  for (int t = 0; t < 4; ++t) bv[t] = bias ? bias[tn + t * 16 + n] : 0.0f;

#pragma unroll
  for (int mi = 0; mi < 2; ++mi) {
#pragma unroll
    for (int r = 0; r < 8; ++r) {
      int row = tm + mi * 16 + g * 8 + r;
#pragma unroll
      for (int t = 0; t < 4; ++t) {
        float v = acc[mi][t][r] + bv[t];
        if (relu) v = fmaxf(v, 0.0f);
        size_t o = (size_t)row * Ndim + tn + t * 16 + n;
        if (Cf) Cf[o] = v;
        if (Cb) Cb[o] = f2bf(v);
      }
    }
  }
}

// ---------------------------------------------------------------- V transpose
// kvb bf16 [M,1024] (cols 512..1023 = V) -> vtb bf16 [B*H][64][L]
__global__ void vtrans_kernel(const unsigned short* __restrict__ kvb,
                              unsigned short* __restrict__ vtb) {
  int bd = blockIdx.x;                 // (b*H + h)*64 + d
  int d  = bd & 63;
  int bh = bd >> 6;
  int h  = bh & 7;
  int b  = bh >> 3;
  for (int l = threadIdx.x; l < L_; l += blockDim.x) {
    unsigned short v = kvb[((size_t)(b * L_ + l)) * (2 * D_) + D_ + h * 64 + d];
    vtb[(size_t)bd * L_ + l] = v;
  }
}

// ---------------------------------------------------------------- fused attention
// one wave per (b, h, 16-row q tile); flash-style online softmax,
// WMMA for Q*K^T and P*V ; ctx written directly as bf16 [M, D]
__global__ __launch_bounds__(32)
void attn_kernel(const unsigned short* __restrict__ qb,   // [M,512]
                 const unsigned short* __restrict__ kvb,  // [M,1024]
                 const unsigned short* __restrict__ vtb,  // [B*H][64][L]
                 unsigned short* __restrict__ ctxb) {     // [M,512]
  __shared__ __align__(16) unsigned short ldsP[16 * 32];

  int lane = threadIdx.x & 31;
  int n = lane & 15;
  int g = lane >> 4;
  int id = blockIdx.x;
  int qt = id % NQT_;
  int bh = id / NQT_;
  int h = bh & 7;
  int b = bh >> 3;

  // Q fragments (16x64 -> two 16x32 A-fragments), row = lane&15
  size_t qr = (size_t)(b * L_ + qt * 16 + n) * D_ + h * 64;
  AF q0, q1;
  q0.q[0] = *(const u32x4*)(qb + qr + g * 16);
  q0.q[1] = *(const u32x4*)(qb + qr + g * 16 + 8);
  q1.q[0] = *(const u32x4*)(qb + qr + 32 + g * 16);
  q1.q[1] = *(const u32x4*)(qb + qr + 32 + g * 16 + 8);

  float mst[8], lst[8];
#pragma unroll
  for (int r = 0; r < 8; ++r) { mst[r] = -3.0e38f; lst[r] = 0.0f; }
  v8f o0 = {}, o1 = {}, o2 = {}, o3 = {};

  const size_t vbase = (size_t)bh * 64 * L_;

  for (int kk = 0; kk < L_; kk += 32) {
    // ---- scores: two 16x16 tiles (keys kk..kk+15, kk+16..kk+31)
    v8f s0 = {}, s1 = {};
    {
      size_t kr = (size_t)(b * L_ + kk + n) * (2 * D_) + h * 64;
      AF ka, kb2;
      ka.q[0]  = *(const u32x4*)(kvb + kr + g * 16);
      ka.q[1]  = *(const u32x4*)(kvb + kr + g * 16 + 8);
      kb2.q[0] = *(const u32x4*)(kvb + kr + 32 + g * 16);
      kb2.q[1] = *(const u32x4*)(kvb + kr + 32 + g * 16 + 8);
      s0 = wmma_bf16(q0, ka, s0);
      s0 = wmma_bf16(q1, kb2, s0);
    }
    int k1 = kk + 16 + n;
    int k1c = k1 < L_ ? k1 : (L_ - 1);       // clamp address, mask value below
    {
      size_t kr = (size_t)(b * L_ + k1c) * (2 * D_) + h * 64;
      AF ka, kb2;
      ka.q[0]  = *(const u32x4*)(kvb + kr + g * 16);
      ka.q[1]  = *(const u32x4*)(kvb + kr + g * 16 + 8);
      kb2.q[0] = *(const u32x4*)(kvb + kr + 32 + g * 16);
      kb2.q[1] = *(const u32x4*)(kvb + kr + 32 + g * 16 + 8);
      s1 = wmma_bf16(q0, ka, s1);
      s1 = wmma_bf16(q1, kb2, s1);
    }
    bool valid1 = (k1 < L_);
#pragma unroll
    for (int r = 0; r < 8; ++r) {
      s0[r] *= 0.125f;                               // 1/sqrt(64)
      s1[r] = valid1 ? (s1[r] * 0.125f) : -3.0e38f;  // mask tail keys
    }

    // ---- online softmax (rows live across 16 lanes with same g)
    float nm[8];
#pragma unroll
    for (int r = 0; r < 8; ++r) nm[r] = fmaxf(mst[r], fmaxf(s0[r], s1[r]));
#pragma unroll
    for (int off = 1; off <= 8; off <<= 1)
#pragma unroll
      for (int r = 0; r < 8; ++r) nm[r] = fmaxf(nm[r], __shfl_xor(nm[r], off));

    float al[8], p0[8], p1[8], rs[8];
#pragma unroll
    for (int r = 0; r < 8; ++r) {
      al[r] = __expf(mst[r] - nm[r]);
      p0[r] = __expf(s0[r] - nm[r]);
      p1[r] = __expf(s1[r] - nm[r]);
      rs[r] = p0[r] + p1[r];
      mst[r] = nm[r];
    }
#pragma unroll
    for (int off = 1; off <= 8; off <<= 1)
#pragma unroll
      for (int r = 0; r < 8; ++r) rs[r] += __shfl_xor(rs[r], off);
#pragma unroll
    for (int r = 0; r < 8; ++r) {
      lst[r] = lst[r] * al[r] + rs[r];
      o0[r] *= al[r]; o1[r] *= al[r]; o2[r] *= al[r]; o3[r] *= al[r];
    }

    // ---- stage P (bf16) through LDS: C-layout -> A-fragment layout
#pragma unroll
    for (int r = 0; r < 8; ++r) {
      int row = g * 8 + r;
      ldsP[row * 32 + n]      = f2bf(p0[r]);
      ldsP[row * 32 + 16 + n] = f2bf(p1[r]);
    }
    __syncthreads();
    AF pf;
    pf.q[0] = *(const u32x4*)(&ldsP[n * 32 + g * 16]);
    pf.q[1] = *(const u32x4*)(&ldsP[n * 32 + g * 16 + 8]);
    __syncthreads();

    // ---- O += P(16x32) x V(32x64), V pre-transposed -> contiguous B frags
    int koff = kk + g * 16;
    if (koff > L_ - 16) koff = L_ - 16;   // clamp (masked keys: P == 0)
#pragma unroll
    for (int t = 0; t < 4; ++t) {
      const unsigned short* vp = vtb + vbase + (size_t)(t * 16 + n) * L_ + koff;
      AF vf;
      vf.q[0] = *(const u32x4*)(vp);
      vf.q[1] = *(const u32x4*)(vp + 8);
      if (t == 0) o0 = wmma_bf16(pf, vf, o0);
      else if (t == 1) o1 = wmma_bf16(pf, vf, o1);
      else if (t == 2) o2 = wmma_bf16(pf, vf, o2);
      else o3 = wmma_bf16(pf, vf, o3);
    }
  }

  // ---- normalize and store ctx (bf16)
#pragma unroll
  for (int r = 0; r < 8; ++r) {
    float inv = 1.0f / lst[r];
    size_t row = (size_t)(b * L_ + qt * 16 + g * 8 + r);
    unsigned short* cp = ctxb + row * D_ + h * 64;
    cp[n]      = f2bf(o0[r] * inv);
    cp[16 + n] = f2bf(o1[r] * inv);
    cp[32 + n] = f2bf(o2[r] * inv);
    cp[48 + n] = f2bf(o3[r] * inv);
  }
}

// ---------------------------------------------------------------- residual + LN
// one wave per row: x = LN(x + add) ; writes f32 and bf16
__global__ __launch_bounds__(32)
void ln_kernel(float* __restrict__ xf, const float* __restrict__ addf,
               const float* __restrict__ gma, const float* __restrict__ bta,
               unsigned short* __restrict__ xb) {
  int m = blockIdx.x;
  int lane = threadIdx.x;
  float vals[16];
  float s = 0.0f;
#pragma unroll
  for (int j = 0; j < 16; ++j) {
    int d = lane + 32 * j;
    vals[j] = xf[(size_t)m * D_ + d] + addf[(size_t)m * D_ + d];
    s += vals[j];
  }
#pragma unroll
  for (int off = 16; off >= 1; off >>= 1) s += __shfl_xor(s, off);
  float mu = s * (1.0f / D_);
  float vs = 0.0f;
#pragma unroll
  for (int j = 0; j < 16; ++j) { float t = vals[j] - mu; vs += t * t; }
#pragma unroll
  for (int off = 16; off >= 1; off >>= 1) vs += __shfl_xor(vs, off);
  float rstd = rsqrtf(vs * (1.0f / D_) + 1e-5f);
#pragma unroll
  for (int j = 0; j < 16; ++j) {
    int d = lane + 32 * j;
    float y = (vals[j] - mu) * rstd * gma[d] + bta[d];
    xf[(size_t)m * D_ + d] = y;
    xb[(size_t)m * D_ + d] = f2bf(y);
  }
}

// ---------------------------------------------------------------- gather rows l>=C
__global__ void gather_kernel(const unsigned short* __restrict__ xb,
                              unsigned short* __restrict__ xqb) {
  int r = blockIdx.x;            // 0 .. B*T-1
  int b = r / T_;
  int t = r - b * T_;
  const unsigned int* s = (const unsigned int*)(xb + (size_t)(b * L_ + C_ + t) * D_);
  unsigned int* d = (unsigned int*)(xqb + (size_t)r * D_);
  d[threadIdx.x] = s[threadIdx.x];   // 256 threads x 2 bf16
}

// ---------------------------------------------------------------- host
extern "C" void kernel_launch(void* const* d_in, const int* in_sizes, int n_in,
                              void* d_out, int out_size, void* d_ws, size_t ws_size,
                              hipStream_t stream) {
  const int*   token     = (const int*)d_in[0];
  const int*   state     = (const int*)d_in[1];
  const float* prefix    = (const float*)d_in[2];
  const float* emb_table = (const float*)d_in[3];
  const float* b_in      = (const float*)d_in[4];
  const float* pos_emb   = (const float*)d_in[5];
  const float* W_out     = (const float*)d_in[6];
  const float* b_out     = (const float*)d_in[7];
  const float* qkv_w     = (const float*)d_in[8];
  const float* qkv_b     = (const float*)d_in[9];
  const float* out_w     = (const float*)d_in[10];
  const float* out_b     = (const float*)d_in[11];
  const float* lin1_w    = (const float*)d_in[12];
  const float* lin1_b    = (const float*)d_in[13];
  const float* lin2_w    = (const float*)d_in[14];
  const float* lin2_b    = (const float*)d_in[15];
  const float* ln_g      = (const float*)d_in[16];
  const float* ln_b      = (const float*)d_in[17];
  float* logits = (float*)d_out;

  char* ws = (char*)d_ws;
  size_t off = 0;
  auto alloc = [&](size_t elems, size_t esz) -> void* {
    void* p = ws + off;
    off += (elems * esz + 255) & ~(size_t)255;
    return p;
  };
  unsigned short* wqkv_b16 = (unsigned short*)alloc((size_t)NL_ * 2 * 3 * D_ * D_, 2);
  unsigned short* woutp_b16= (unsigned short*)alloc((size_t)NL_ * 2 * D_ * D_, 2);
  unsigned short* wlin1_b16= (unsigned short*)alloc((size_t)NL_ * DFF_ * D_, 2);
  unsigned short* wlin2_b16= (unsigned short*)alloc((size_t)NL_ * D_ * DFF_, 2);
  unsigned short* wout_b16 = (unsigned short*)alloc((size_t)V_ * S_ * D_, 2);
  float*          xf   = (float*)alloc((size_t)M_ * D_, 4);
  unsigned short* xb   = (unsigned short*)alloc((size_t)M_ * D_, 2);
  unsigned short* memb = (unsigned short*)alloc((size_t)M_ * D_, 2);
  unsigned short* qb   = (unsigned short*)alloc((size_t)M_ * D_, 2);
  unsigned short* kvb  = (unsigned short*)alloc((size_t)M_ * 2 * D_, 2);
  unsigned short* vtb  = (unsigned short*)alloc((size_t)B_ * H_ * 64 * L_, 2);
  unsigned short* ctxb = (unsigned short*)alloc((size_t)M_ * D_, 2);
  float*          tmpf = (float*)alloc((size_t)M_ * D_, 4);
  unsigned short* hb   = (unsigned short*)alloc((size_t)M_ * DFF_, 2);
  unsigned short* xqb  = (unsigned short*)alloc((size_t)B_ * T_ * D_, 2);

  auto conv = [&](const float* s, unsigned short* d, long long n) {
    long long blocks = (n + 1023) / 1024;
    if (blocks > 4096) blocks = 4096;
    f32_to_bf16_kernel<<<dim3((unsigned)blocks), 256, 0, stream>>>(s, d, n);
  };
  auto gemm = [&](const unsigned short* A, const unsigned short* Wm, const float* bias,
                  float* Cf, unsigned short* Cb, int Mdim, int Ndim, int Kdim, int relu) {
    dim3 g(Ndim / 64, Mdim / 128);
    gemm_kernel<<<g, 128, 0, stream>>>(A, Wm, bias, Cf, Cb, Mdim, Ndim, Kdim, relu);
  };

  // weights -> bf16
  conv(qkv_w,  wqkv_b16,  (long long)NL_ * 2 * 3 * D_ * D_);
  conv(out_w,  woutp_b16, (long long)NL_ * 2 * D_ * D_);
  conv(lin1_w, wlin1_b16, (long long)NL_ * DFF_ * D_);
  conv(lin2_w, wlin2_b16, (long long)NL_ * D_ * DFF_);
  conv(W_out,  wout_b16,  (long long)V_ * S_ * D_);

  // embedding + prefix + pos
  embed_kernel<<<M_, 256, 0, stream>>>(token, state, prefix, emb_table, b_in,
                                       pos_emb, xf, xb, memb);

  for (int i = 0; i < NL_; ++i) {
    for (int j = 0; j < 2; ++j) {                 // j=0 self-attn, j=1 cross-attn
      const unsigned short* kvA = (j == 0) ? xb : memb;
      const unsigned short* wq  = wqkv_b16 + (size_t)(i * 2 + j) * 3 * D_ * D_;
      const unsigned short* wkv = wq + (size_t)D_ * D_;
      const float* bq  = qkv_b + (size_t)(i * 2 + j) * 3 * D_;
      const float* bkv = bq + D_;
      gemm(xb,  wq,  bq,  nullptr, qb,  M_, D_,     D_, 0);   // Q
      gemm(kvA, wkv, bkv, nullptr, kvb, M_, 2 * D_, D_, 0);   // K|V
      vtrans_kernel<<<B_ * H_ * 64, 256, 0, stream>>>(kvb, vtb);
      attn_kernel<<<B_ * H_ * NQT_, 32, 0, stream>>>(qb, kvb, vtb, ctxb);
      gemm(ctxb, woutp_b16 + (size_t)(i * 2 + j) * D_ * D_,
           out_b + (size_t)(i * 2 + j) * D_, tmpf, nullptr, M_, D_, D_, 0);
      ln_kernel<<<M_, 32, 0, stream>>>(xf, tmpf, ln_g + (size_t)(i * 3 + j) * D_,
                                       ln_b + (size_t)(i * 3 + j) * D_, xb);
    }
    // FFN
    gemm(xb, wlin1_b16 + (size_t)i * DFF_ * D_, lin1_b + (size_t)i * DFF_,
         nullptr, hb, M_, DFF_, D_, 1);
    gemm(hb, wlin2_b16 + (size_t)i * D_ * DFF_, lin2_b + (size_t)i * D_,
         tmpf, nullptr, M_, D_, DFF_, 0);
    ln_kernel<<<M_, 32, 0, stream>>>(xf, tmpf, ln_g + (size_t)(i * 3 + 2) * D_,
                                     ln_b + (size_t)(i * 3 + 2) * D_, xb);
  }

  // drop prefix rows, final projection straight into d_out [B*T, V*S] f32
  gather_kernel<<<B_ * T_, 256, 0, stream>>>(xb, xqb);
  gemm(xqb, wout_b16, b_out, logits, nullptr, B_ * T_, V_ * S_, D_, 0);
}
